// FlowPointContrastWaymo_4810363372455
// MI455X (gfx1250) — matrix operand order repaired and checked by reading, not compiled, not run on hardware
//
#include <hip/hip_runtime.h>
#include <hip/hip_bf16.h>

typedef __attribute__((ext_vector_type(16))) _Float16 v16h;
typedef __attribute__((ext_vector_type(8)))  float    v8f;

#define TEMP_INV   14.285714285714286f   /* 1/0.07 */
#define THRESH     0.8f
#define EPS_NORM   1e-7f
#define EPS_CLAMP  1e-6f
#define D_FEAT     256

union HU { float4 f4[2]; v16h h; };

// ---------------------------------------------------------------------------
// Kernel 0: zero the global accumulators (num[B], den[B]) in workspace.
// ---------------------------------------------------------------------------
__global__ void fpc_init(float* __restrict__ acc, int n) {
    int t = threadIdx.x;
    if (t < n) acc[t] = 0.0f;
}

// ---------------------------------------------------------------------------
// Kernel 1: L2-normalize each feature row (D=256) and store as f16.
// One 256-thread block per row; wave32 shuffle reduce + LDS combine.
// ---------------------------------------------------------------------------
__global__ __launch_bounds__(256)
void fpc_norm(const float* __restrict__ f, _Float16* __restrict__ nf) {
    const size_t row = blockIdx.x;
    const int tid = threadIdx.x;
    float v = f[row * D_FEAT + tid];
    float ss = v * v;
    #pragma unroll
    for (int off = 16; off > 0; off >>= 1)
        ss += __shfl_xor(ss, off, 32);
    __shared__ float part[8];
    __shared__ float sinv;
    if ((tid & 31) == 0) part[tid >> 5] = ss;
    __syncthreads();
    if (tid == 0) {
        float t = 0.0f;
        #pragma unroll
        for (int i = 0; i < 8; ++i) t += part[i];
        sinv = 1.0f / (sqrtf(t) + EPS_NORM);
    }
    __syncthreads();
    nf[row * D_FEAT + tid] = (_Float16)(v * sinv);
}

// ---------------------------------------------------------------------------
// Kernel 2: fused Gram-matrix + contrastive-loss epilogue.
// Block = 256 threads (8 wave32) owns rows [i0, i0+32) of batch b: each wave
// keeps TWO 16-row A slabs in VGPRs and runs two WMMA accumulation chains per
// B tile (B loaded once, used twice). Waves partition the column tiles.
// Two passes over all N columns; logits recomputed with v_wmma f16->f32.
// sim_scores streamed with non-temporal loads so nf stays L2-resident.
// Templated on N so sim-tile loads use one base + immediate offsets
// (compile-time v*N strides) instead of per-load 64-bit address adds.
// ---------------------------------------------------------------------------
template <int NC>
__global__ __launch_bounds__(256)
void fpc_main(const _Float16* __restrict__ nf, const float* __restrict__ sim,
              float* __restrict__ num, float* __restrict__ den, int nDyn) {
    const int N    = (NC > 0) ? NC : nDyn;
    const int b    = blockIdx.y;
    const int i0   = blockIdx.x << 5;    // 32 rows per block
    const int tid  = threadIdx.x;
    const int lane = tid & 31;
    const int wave = tid >> 5;           // 0..7
    const int g    = lane >> 4;          // half-wave select
    const int ln   = lane & 15;          // tile row/col within half

    __shared__ float s_neg[32];
    __shared__ float s_pos[32];
    __shared__ float s_nll[32];
    if (tid < 32) { s_neg[tid] = 0.0f; s_pos[tid] = 0.0f; s_nll[tid] = 0.0f; }
    __syncthreads();

    const _Float16* fb = nf  + (size_t)b * N * D_FEAT;
    const float*    sb = sim + (size_t)b * N * N;

    // A operands: two 16-row slabs, all of K=256, resident in VGPRs.
    // Layout used here (16-bit A 16x32): lane ln holds row m=ln;
// halves 0..15 -> K = 32*kk + 16*g + h (contiguous, matching B).
    v16h A[2][8];
    #pragma unroll
    for (int sl = 0; sl < 2; ++sl) {
        const _Float16* rowA = fb + (size_t)(i0 + 16 * sl + ln) * D_FEAT;
        #pragma unroll
        for (int kk = 0; kk < 8; ++kk) {
            HU u;
            u.f4[0] = *(const float4*)(rowA + 32 * kk + 16 * g);
            u.f4[1] = *(const float4*)(rowA + 32 * kk + 16 * g + 8);
            A[sl][kk] = u.h;
        }
    }

    const int NT = N >> 4;               // column tiles
    const int NW = 8;                    // waves per block

    // ---------------- Pass 1: negsum_i and poscnt_i --------------------------
    float accNeg[2][8], accPos[2][8];
    #pragma unroll
    for (int sl = 0; sl < 2; ++sl)
        #pragma unroll
        for (int v = 0; v < 8; ++v) { accNeg[sl][v] = 0.0f; accPos[sl][v] = 0.0f; }

    for (int jt = wave; jt < NT; jt += NW) {
        const int j0 = jt << 4;
        const _Float16* rowB = fb + (size_t)(j0 + ln) * D_FEAT;
        {   // prefetch next tile's B row
            int jn = jt + NW;
            if (jn < NT)
                __builtin_prefetch(fb + (size_t)((jn << 4) + ln) * D_FEAT, 0, 0);
        }
        v8f c0 = {}, c1 = {};
        #pragma unroll
        for (int kk = 0; kk < 8; ++kk) {
            // B operand (32x16, K x N): lane ln = column j0+ln; 16 contiguous
            // halves starting at K = 32*kk + 16*g (same k order as A). One load,
            // two WMMA chains.
            HU u;
            u.f4[0] = *(const float4*)(rowB + 32 * kk + 16 * g);
            u.f4[1] = *(const float4*)(rowB + 32 * kk + 16 * g + 8);
            c0 = __builtin_amdgcn_wmma_f32_16x16x32_f16(false, A[0][kk], false, u.h,
                                                        (short)0, c0, false, false);
            c1 = __builtin_amdgcn_wmma_f32_16x16x32_f16(false, A[1][kk], false, u.h,
                                                        (short)0, c1, false, false);
        }
        // C/D layout: lane -> n = ln, VGPR v -> m = v + 8*g (+16 for slab 1).
        // With NC constexpr, v*N strides are immediates: one base, 16 offsets.
        const float* srow = sb + (size_t)(i0 + 8 * g) * N + (j0 + ln);
        #pragma unroll
        for (int v = 0; v < 8; ++v) {
            float s0 = __builtin_nontemporal_load(srow + (size_t)v * N);
            float s1 = __builtin_nontemporal_load(srow + (size_t)(v + 16) * N);
            float l0 = c0[v] * TEMP_INV;
            float l1 = c1[v] * TEMP_INV;
            float e0 = __expf(l0);
            float e1 = __expf(l1);
            bool  p0 = s0 > THRESH;
            bool  p1 = s1 > THRESH;
            accNeg[0][v] += p0 ? 0.0f : e0;
            accPos[0][v] += p0 ? 1.0f : 0.0f;
            accNeg[1][v] += p1 ? 0.0f : e1;
            accPos[1][v] += p1 ? 1.0f : 0.0f;
        }
    }
    #pragma unroll
    for (int sl = 0; sl < 2; ++sl)
        #pragma unroll
        for (int v = 0; v < 8; ++v) {
            atomicAdd(&s_neg[16 * sl + v + 8 * g], accNeg[sl][v]);
            atomicAdd(&s_pos[16 * sl + v + 8 * g], accPos[sl][v]);
        }
    __syncthreads();

    // ---------------- Pass 2: weighted nll sums ------------------------------
    float negRow[2][8];
    #pragma unroll
    for (int sl = 0; sl < 2; ++sl)
        #pragma unroll
        for (int v = 0; v < 8; ++v) negRow[sl][v] = s_neg[16 * sl + v + 8 * g];

    float accNll[2][8];
    #pragma unroll
    for (int sl = 0; sl < 2; ++sl)
        #pragma unroll
        for (int v = 0; v < 8; ++v) accNll[sl][v] = 0.0f;

    for (int jt = wave; jt < NT; jt += NW) {
        const int j0 = jt << 4;
        const _Float16* rowB = fb + (size_t)(j0 + ln) * D_FEAT;
        v8f c0 = {}, c1 = {};
        #pragma unroll
        for (int kk = 0; kk < 8; ++kk) {
            HU u;
            u.f4[0] = *(const float4*)(rowB + 32 * kk + 16 * g);
            u.f4[1] = *(const float4*)(rowB + 32 * kk + 16 * g + 8);
            c0 = __builtin_amdgcn_wmma_f32_16x16x32_f16(false, A[0][kk], false, u.h,
                                                        (short)0, c0, false, false);
            c1 = __builtin_amdgcn_wmma_f32_16x16x32_f16(false, A[1][kk], false, u.h,
                                                        (short)0, c1, false, false);
        }
        const float* srow = sb + (size_t)(i0 + 8 * g) * N + (j0 + ln);
        #pragma unroll
        for (int v = 0; v < 8; ++v) {
            float s0 = __builtin_nontemporal_load(srow + (size_t)v * N);
            float s1 = __builtin_nontemporal_load(srow + (size_t)(v + 16) * N);
            float l0 = c0[v] * TEMP_INV;
            float l1 = c1[v] * TEMP_INV;
            if (s0 > THRESH)
                accNll[0][v] += __logf(__expf(l0) + negRow[0][v]) - l0;
            if (s1 > THRESH)
                accNll[1][v] += __logf(__expf(l1) + negRow[1][v]) - l1;
        }
    }
    #pragma unroll
    for (int sl = 0; sl < 2; ++sl)
        #pragma unroll
        for (int v = 0; v < 8; ++v)
            atomicAdd(&s_nll[16 * sl + v + 8 * g], accNll[sl][v]);
    __syncthreads();

    // ---------------- Row epilogue: valid rows into global accumulators ------
    if (tid < 32) {
        float pc = s_pos[tid];
        if (pc > 0.0f) {
            float row = s_nll[tid] / fmaxf(pc, EPS_CLAMP);
            atomicAdd(&num[b], row);
            atomicAdd(&den[b], 1.0f);
        }
    }
}

// ---------------------------------------------------------------------------
// Kernel 3: final scalar = mean_b( num[b] / max(den[b], eps) ).
// ---------------------------------------------------------------------------
__global__ void fpc_fin(const float* __restrict__ num, const float* __restrict__ den,
                        float* __restrict__ out, int B) {
    if (threadIdx.x == 0 && blockIdx.x == 0) {
        float t = 0.0f;
        for (int b = 0; b < B; ++b) t += num[b] / fmaxf(den[b], EPS_CLAMP);
        out[0] = t / (float)B;
    }
}

// ---------------------------------------------------------------------------
extern "C" void kernel_launch(void* const* d_in, const int* in_sizes, int n_in,
                              void* d_out, int out_size, void* d_ws, size_t ws_size,
                              hipStream_t stream) {
    const float* feat = (const float*)d_in[0];   // [B, N, D] f32
    const float* sim  = (const float*)d_in[1];   // [B, N, N] f32
    float* out = (float*)d_out;                  // scalar f32

    const long long BN = (long long)in_sizes[0] / D_FEAT;   // B*N
    const int N = (int)((long long)in_sizes[1] / BN);       // N
    const int B = (int)(BN / N);                            // B

    // workspace layout: [num[B] den[B]] @ 0, f16 normalized features @ +256B
    float* acc = (float*)d_ws;
    _Float16* nf = (_Float16*)((char*)d_ws + 256);

    fpc_init<<<1, 32, 0, stream>>>(acc, 2 * B);
    fpc_norm<<<dim3((unsigned)BN), 256, 0, stream>>>(feat, nf);
    if (N == 4096)
        fpc_main<4096><<<dim3(N / 32, B), 256, 0, stream>>>(nf, sim, acc, acc + B, N);
    else
        fpc_main<0><<<dim3(N / 32, B), 256, 0, stream>>>(nf, sim, acc, acc + B, N);
    fpc_fin<<<1, 32, 0, stream>>>(acc, acc + B, out, B);
}
